// Window_Loss2_63505386438799
// MI455X (gfx1250) — matrix operand order repaired and checked by reading, not compiled, not run on hardware
//
#include <hip/hip_runtime.h>

#define NW 16
#define NG 5
#define NF 10
#define NPAIR 15      // NG+NF
#define STEP 16
#define T 256
#define DF 64
#define NSEQ 256      // NW*STEP
#define NPAIRS 240    // NW*NPAIR
#define GAMMA 5.0f
#define INV_GAMMA 0.2f
#define MARGIN 1.0f
#define PCOEF 0.5f
#define RCOEF 0.01f
#define INFV 1.0e9f

typedef __attribute__((ext_vector_type(16))) __bf16 v16bf;
typedef __attribute__((ext_vector_type(8)))  float  v8f;

// ---------------------------------------------------------------------------
// Kernel 1: per-timestep squared norms: norms[s*T+t] = sum_k data[s][t][k]^2
// ---------------------------------------------------------------------------
__global__ void __launch_bounds__(256)
norms_kernel(const float* __restrict__ data, float* __restrict__ norms) {
  int s = blockIdx.x;
  int t = threadIdx.x;
  const float4* p = (const float4*)(data + ((size_t)s * T + t) * DF);
  float acc = 0.f;
#pragma unroll
  for (int q = 0; q < DF / 4; ++q) {
    float4 v = p[q];
    acc += v.x * v.x + v.y * v.y + v.z * v.z + v.w * v.w;
  }
  norms[s * T + t] = acc;
}

// Convert 8 contiguous f32 to bf16 into dst[base..base+7]
__device__ __forceinline__ void cvt8(const float* __restrict__ src, v16bf& dst, int base) {
  float4 f0 = *(const float4*)(src);
  float4 f1 = *(const float4*)(src + 4);
  dst[base + 0] = (__bf16)f0.x; dst[base + 1] = (__bf16)f0.y;
  dst[base + 2] = (__bf16)f0.z; dst[base + 3] = (__bf16)f0.w;
  dst[base + 4] = (__bf16)f1.x; dst[base + 5] = (__bf16)f1.y;
  dst[base + 6] = (__bf16)f1.z; dst[base + 7] = (__bf16)f1.w;
}

// ---------------------------------------------------------------------------
// Kernel 2: per-pair cost matrix (bf16 WMMA -> f32, resident in LDS) + soft-DTW
// One workgroup (8 wave32) per pair. Dynamic LDS:
//   sD[256*256] | nX[256] | nY[256] | 3 DP diagonal buffers of (T+1) w/ guard
// ---------------------------------------------------------------------------
__global__ void __launch_bounds__(256, 1)
pair_dtw_kernel(const float* __restrict__ data, const int* __restrict__ lens,
                const float* __restrict__ norms, float* __restrict__ dout) {
  extern __shared__ float smem[];
  float* sD     = smem;                 // 65536 floats
  float* nX     = smem + T * T;         // 256
  float* nY     = nX + T;               // 256
  float* rowraw = nY + T;               // 3*(T+1); index 0 of each = guard (-1)

  const int p  = blockIdx.x;
  const int w  = p / NPAIR;
  const int o  = p - w * NPAIR;
  const int sx = w * STEP;              // anchor sequence
  const int sy = sx + 1 + o;            // other sequence
  const float* X = data + (size_t)sx * T * DF;
  const float* Y = data + (size_t)sy * T * DF;

  const int tid  = threadIdx.x;
  const int lane = tid & 31;
  const int wv   = tid >> 5;
  const int lrow = lane & 15;
  const int lhi  = (lane >> 4) & 1;

  // stage row norms + init DP buffers (guards included) to INF
  nX[tid] = norms[sx * T + tid];
  nY[tid] = norms[sy * T + tid];
#pragma unroll
  for (int q = tid; q < 3 * (T + 1); q += 256) rowraw[q] = INFV;
  __syncthreads();

  // ---- GEMM phase: each wave owns 2 row-tiles x 16 col-tiles ----
  for (int rt = wv * 2; rt < wv * 2 + 2; ++rt) {
    const int m0 = rt * 16;
    // A fragment (16x32 bf16): lanes 0-15 -> M=lane,    K = k0+[0..7],[16..23]
    //                          lanes 16-31 -> M=lane-16, K = k0+[8..15],[24..31]
    const float* xr = X + (size_t)(m0 + lrow) * DF;
    v16bf a0, a1;
    {
      int ka = lhi * 8;                 // k0 = 0
      cvt8(xr + ka,      a0, 0);
      cvt8(xr + ka + 16, a0, 8);
      ka = 32 + lhi * 8;                // k0 = 32
      cvt8(xr + ka,      a1, 0);
      cvt8(xr + ka + 16, a1, 8);
    }
    // hoist anchor norms for this wave's 8 output rows into VGPRs
    float nxr[8];
#pragma unroll
    for (int r = 0; r < 8; ++r) nxr[r] = nX[m0 + 8 * lhi + r];
    float* sdst = sD + (size_t)(m0 + 8 * lhi) * T + lrow;

    for (int ct = 0; ct < 16; ++ct) {
      const int j0 = ct * 16;
      // B fragment (32x16 bf16): lane holds column N=lane%16 = Y row j0+N;
      // lanes 0-15: K = k0+[0..7],[16..23]; lanes 16-31: K = k0+[8..15],[24..31].
      const float* yr = Y + (size_t)(j0 + lrow) * DF;
      if (ct < 15) __builtin_prefetch(yr + 16 * DF, 0, 1);   // next B tile
      v16bf b0, b1;
      const int kb = lhi * 8;
      cvt8(yr + kb,          b0, 0);
      cvt8(yr + kb + 16,     b0, 8);
      cvt8(yr + 32 + kb,     b1, 0);
      cvt8(yr + 32 + kb + 16, b1, 8);

      v8f c = {0.f, 0.f, 0.f, 0.f, 0.f, 0.f, 0.f, 0.f};
      c = __builtin_amdgcn_wmma_f32_16x16x32_bf16(false, a0, false, b0,
                                                  (short)0, c, false, false);
      c = __builtin_amdgcn_wmma_f32_16x16x32_bf16(false, a1, false, b1,
                                                  (short)0, c, false, false);

      // D[i][j] = |x_i|^2 + |y_j|^2 - 2 x_i.y_j   (C layout: M=r+8*lhi, N=lane%16)
      const float ny = nY[j0 + lrow];
#pragma unroll
      for (int r = 0; r < 8; ++r) {
        sdst[r * T + j0] = nxr[r] + ny - 2.0f * c[r];
      }
    }
  }

  // ---- soft-DTW wavefront phase ----
  const int la  = lens[sx];
  const int lb  = lens[sy];
  const int tgt = la + lb - 2;
  const float invLen = 1.0f / (float)(la + lb);

  // logical DP row pointers (element -1 = permanent INF guard)
  float* B0 = rowraw + 1;
  float* B1 = rowraw + (T + 1) + 1;
  float* B2 = rowraw + 2 * (T + 1) + 1;

  const int i = tid;
  const float* pD = sD + (size_t)i * (T - 1);   // pD[k] == D[i][k-i]

  auto step = [&](int k, float* cur, const float* prev, const float* prev2) {
    float val = INFV;
    if ((unsigned)(k - i) < (unsigned)T) {      // j = k-i in [0, T)
      const float Dk   = pD[k];
      const float up   = prev[i - 1];           // guard gives INF at i=0
      const float left = prev[i];
      const float diag = prev2[i - 1];          // guard gives INF at i=0 (k>=1)
      const float mn = fminf(up, fminf(left, diag));
      const float s = __expf((mn - up) * INV_GAMMA) +
                      __expf((mn - left) * INV_GAMMA) +
                      __expf((mn - diag) * INV_GAMMA);
      val = Dk + mn - GAMMA * __logf(s);
    }
    cur[i] = val;
    if (k == tgt && i == la - 1) dout[p] = val * invLen;
    __syncthreads();
  };

  // peeled k=0: R[0,0] = D[0,0] (softmin(INF,INF,0)=0); tgt >= 126 so no extract
  B0[i] = (i == 0) ? sD[0] : INFV;
  __syncthreads();

  // k = 1..510, statically rotated buffers (510 = 3*170)
  for (int k = 1; k < 2 * T - 1; k += 3) {
    step(k,     B1, B0, B2);
    step(k + 1, B2, B1, B0);
    step(k + 2, B0, B2, B1);
  }
}

// ---------------------------------------------------------------------------
// Kernel 3: finalize — triplet loss per window + softmax window loss
// ---------------------------------------------------------------------------
__global__ void finalize_kernel(const float* __restrict__ dvals, float* __restrict__ out) {
  if (threadIdx.x != 0 || blockIdx.x != 0) return;

  // total_loss = mean over windows of sum(relu(dg+M-dn)*0.5)/(cnt+1)
  float total = 0.f;
  for (int w = 0; w < NW; ++w) {
    const float* dw = dvals + w * NPAIR;
    float ssum = 0.f;
    int cnt = 0;
    for (int g = 0; g < NG; ++g) {
      const float dg = dw[g];
      for (int f = 0; f < NF; ++f) {
        const float v = dg + MARGIN - dw[NG + f];
        if (v > 0.f) { ssum += v * 0.5f; cnt++; }   // P = 1-P = 0.5
      }
    }
    total += ssum / (float)(cnt + 1);
  }
  total *= (1.0f / (float)NW);

  // dp = sum(dg * softmax(R*dg)) over 80 genuine distances
  float mx = -1e30f;
  for (int w = 0; w < NW; ++w)
    for (int g = 0; g < NG; ++g)
      mx = fmaxf(mx, RCOEF * dvals[w * NPAIR + g]);
  float se = 0.f, sv = 0.f;
  for (int w = 0; w < NW; ++w)
    for (int g = 0; g < NG; ++g) {
      const float d = dvals[w * NPAIR + g];
      const float e = __expf(RCOEF * d - mx);
      se += e; sv += d * e;
    }
  const float dp = sv / se;

  // dnv = sum(dn * softmax(-R*dn)) over 160 forgery distances
  float mx2 = -1e30f;
  for (int w = 0; w < NW; ++w)
    for (int f = 0; f < NF; ++f)
      mx2 = fmaxf(mx2, -RCOEF * dvals[w * NPAIR + NG + f]);
  float se2 = 0.f, sv2 = 0.f;
  for (int w = 0; w < NW; ++w)
    for (int f = 0; f < NF; ++f) {
      const float d = dvals[w * NPAIR + NG + f];
      const float e = __expf(-RCOEF * d - mx2);
      se2 += e; sv2 += d * e;
    }
  const float dnv = sv2 / se2;

  const float wl = fmaxf(dp - dnv, 0.f);
  out[0] = wl * PCOEF + total * (1.0f - PCOEF);
}

// ---------------------------------------------------------------------------
extern "C" void kernel_launch(void* const* d_in, const int* in_sizes, int n_in,
                              void* d_out, int out_size, void* d_ws, size_t ws_size,
                              hipStream_t stream) {
  const float* data = (const float*)d_in[0];   // [256, 256, 64] f32
  const int*   lens = (const int*)d_in[1];     // [256] i32

  float* norms = (float*)d_ws;                 // 65536 f32
  float* dvals = norms + NSEQ * T;             // 240 f32

  norms_kernel<<<NSEQ, T, 0, stream>>>(data, norms);

  const size_t shmem =
      (size_t)(T * T + 2 * T + 3 * (T + 1)) * sizeof(float);  // 267,276 B
  (void)hipFuncSetAttribute((const void*)pair_dtw_kernel,
                            hipFuncAttributeMaxDynamicSharedMemorySize,
                            (int)shmem);
  pair_dtw_kernel<<<NPAIRS, 256, shmem, stream>>>(data, lens, norms, dvals);

  finalize_kernel<<<1, 32, 0, stream>>>(dvals, (float*)d_out);
}
